// RNN_model_7937099563067
// MI455X (gfx1250) — matrix operand order repaired and checked by reading, not compiled, not run on hardware
//
#include <hip/hip_runtime.h>

// ---------------------------------------------------------------------------
// Types for CDNA5 WMMA (wave32, v_wmma_f32_16x16x32_bf16)
// ---------------------------------------------------------------------------
typedef __bf16 v16bf __attribute__((ext_vector_type(16)));
typedef __bf16 v8bf  __attribute__((ext_vector_type(8)));
typedef float  v8f   __attribute__((ext_vector_type(8)));

union Frag {
    v16bf v;
    v8bf  h[2];
};

__device__ __forceinline__ __bf16 f2bf(float f) {
    union { float f; unsigned u; } x; x.f = f;
    unsigned r = x.u + 0x7FFFu + ((x.u >> 16) & 1u);   // round-to-nearest-even
    unsigned short hs = (unsigned short)(r >> 16);
    return __builtin_bit_cast(__bf16, hs);
}

__device__ __forceinline__ float sigmoidf_(float x) {
    return 1.0f / (1.0f + __expf(-x));
}

__device__ __forceinline__ v8f wmma_bf16(const Frag& a, const Frag& b, v8f c) {
    return __builtin_amdgcn_wmma_f32_16x16x32_bf16(
        false, a.v, false, b.v, (short)0, c, false, false);
}

// CDNA5 async global->LDS copy, 16B per lane, tracked by ASYNCcnt.
// lds = low 32 bits of a flat pointer to __shared__ (== wave LDS offset).
__device__ __forceinline__ void async_b128(unsigned lds, const void* g) {
    asm volatile("global_load_async_to_lds_b128 %0, %1, off"
                 :: "v"(lds), "v"(g) : "memory");
}
#define S_WAIT_ASYNC(n) asm volatile("s_wait_asynccnt %0" :: "n"(n) : "memory")

__device__ __forceinline__ unsigned lds_addr(const void* p) {
    return (unsigned)(unsigned long long)p;
}

// ---------------------------------------------------------------------------
// Problem constants
// ---------------------------------------------------------------------------
#define BB 8192
#define FF 512
#define EE 512
#define KK 1024     // F + E == U
#define UU 1024
#define VV 12000
#define LDA 40      // padded LDS row stride (elements): conflict-free A-frag reads

// ---------------------------------------------------------------------------
// K1: gather emb rows, concat with features, convert to bf16
// ---------------------------------------------------------------------------
__global__ void __launch_bounds__(256) prep_x_kernel(
    const float* __restrict__ features, const float* __restrict__ emb,
    const int* __restrict__ S, __bf16* __restrict__ X)
{
    int tid = blockIdx.x * 256 + threadIdx.x;
    int b = tid >> 10;
    int j = tid & 1023;
    float v = (j < FF) ? features[(size_t)b * FF + j]
                       : emb[(size_t)S[b] * EE + (j - FF)];
    X[tid] = f2bf(v);
}

// ---------------------------------------------------------------------------
// K2: transpose fp32 (R x C, row-major) -> bf16 (C x R, row-major)
// ---------------------------------------------------------------------------
__global__ void __launch_bounds__(256) transpose_f32_bf16_kernel(
    const float* __restrict__ in, __bf16* __restrict__ out, int R, int C)
{
    __shared__ float tile[32][33];
    int cx = blockIdx.x * 32;
    int ry = blockIdx.y * 32;
#pragma unroll
    for (int i = 0; i < 32; i += 8) {
        int r = ry + threadIdx.y + i;
        tile[threadIdx.y + i][threadIdx.x] = in[(size_t)r * C + cx + threadIdx.x];
    }
    __syncthreads();
#pragma unroll
    for (int i = 0; i < 32; i += 8) {
        int orow = cx + threadIdx.y + i;
        out[(size_t)orow * R + ry + threadIdx.x] = f2bf(tile[threadIdx.x][threadIdx.y + i]);
    }
}

// ---------------------------------------------------------------------------
// K3: fused GEMM1 + LSTM cell (h0 = c0 = 0 -> f-gate and W_r are dead)
//   gates used: i (W_k cols 0..U), g (2U..3U), o (3U..4U)
//   c = sigmoid(i)*g ; h = sigmoid(o)*c
// Block 128 rows x 64 U-cols, 8 waves (4 row x 2 col), 3 gates x 2x2 tiles.
// Triple-buffered LDS fed by global_load_async_to_lds_b128; 1 barrier/k-step.
// ---------------------------------------------------------------------------
__global__ void __launch_bounds__(256) lstm_gemm1_kernel(
    const __bf16* __restrict__ X,    // 8192 x 1024
    const __bf16* __restrict__ Wt,   // 4096 x 1024 (N-major)
    const float*  __restrict__ bl,   // 4096
    float* __restrict__ outH, float* __restrict__ outC,
    __bf16* __restrict__ Hbf)
{
    __shared__ alignas(16) __bf16 As[3 * 128 * LDA];
    __shared__ alignas(16) __bf16 Bs[3 * 192 * LDA];

    const int t    = threadIdx.x;
    const int lane = t & 31;
    const int wid  = t >> 5;
    const int wr   = wid & 3;
    const int wc   = wid >> 2;
    const int lr   = lane & 15;
    const int lh   = lane >> 4;

    const int m0 = blockIdx.x * 128;
    const int u0 = blockIdx.y * 64;

    // --- per-thread staging descriptors ---
    const int ar  = t >> 1;
    const int akc = (t & 1) << 4;
    const __bf16* agp = X + (size_t)(m0 + ar) * KK + akc;
    const unsigned al = lds_addr(&As[ar * LDA + akc]);

    const int br  = t >> 2;               // 0..63 : nn within the u-tile
    const int bkc = (t & 3) << 3;
    const __bf16* bg0 = Wt + (size_t)(0 * UU + u0 + br) * KK + bkc;  // gate i
    const __bf16* bg1 = Wt + (size_t)(2 * UU + u0 + br) * KK + bkc;  // gate g
    const __bf16* bg2 = Wt + (size_t)(3 * UU + u0 + br) * KK + bkc;  // gate o
    const unsigned bl0 = lds_addr(&Bs[(br      ) * LDA + bkc]);
    const unsigned bl1 = lds_addr(&Bs[(br +  64) * LDA + bkc]);
    const unsigned bl2 = lds_addr(&Bs[(br + 128) * LDA + bkc]);

    v8f acc[3][2][2] = {};

    auto stage = [&](int s) {
        const size_t  go = (size_t)s * 32;
        const unsigned ao = (unsigned)((s % 3) * (128 * LDA * 2));
        const unsigned bo = (unsigned)((s % 3) * (192 * LDA * 2));
        async_b128(al + ao,      agp + go);
        async_b128(al + ao + 16, agp + go + 8);
        async_b128(bl0 + bo, bg0 + go);
        async_b128(bl1 + bo, bg1 + go);
        async_b128(bl2 + bo, bg2 + go);
    };

    stage(0);
    const int NS = KK / 32;
    for (int s = 0; s < NS; ++s) {
        if (s + 1 < NS) { stage(s + 1); S_WAIT_ASYNC(5); }
        else            { S_WAIT_ASYNC(0); }
        __syncthreads();

        const __bf16* Ab = As + (s % 3) * (128 * LDA);
        const __bf16* Bb = Bs + (s % 3) * (192 * LDA);

        Frag a[2];
#pragma unroll
        for (int m = 0; m < 2; ++m) {
            const __bf16* p = Ab + (wr * 32 + m * 16 + lr) * LDA + lh * 8;
            a[m].h[0] = *(const v8bf*)p;
            a[m].h[1] = *(const v8bf*)(p + 16);
        }
#pragma unroll
        for (int g = 0; g < 3; ++g) {
#pragma unroll
            for (int n = 0; n < 2; ++n) {
                Frag b;
                const __bf16* q = Bb + (g * 64 + wc * 32 + n * 16 + lr) * LDA + lh * 8;
                b.h[0] = *(const v8bf*)q;
                b.h[1] = *(const v8bf*)(q + 16);
#pragma unroll
                for (int m = 0; m < 2; ++m)
                    acc[g][m][n] = wmma_bf16(a[m], b, acc[g][m][n]);
            }
        }
    }

#pragma unroll
    for (int n = 0; n < 2; ++n) {
        int colU = u0 + wc * 32 + n * 16 + lr;
        float bi = bl[colU];
        float bg = bl[2 * UU + colU];
        float bo = bl[3 * UU + colU];
#pragma unroll
        for (int m = 0; m < 2; ++m) {
            int rbase = m0 + wr * 32 + m * 16 + 8 * lh;
#pragma unroll
            for (int v = 0; v < 8; ++v) {
                int row = rbase + v;
                float iv = sigmoidf_(acc[0][m][n][v] + bi);
                float gv = acc[1][m][n][v] + bg;       // g gate is NOT activated
                float ov = sigmoidf_(acc[2][m][n][v] + bo);
                float cv = iv * gv;
                float hv = ov * cv;
                size_t idx = (size_t)row * UU + colU;
                outC[idx] = cv;
                outH[idx] = hv;
                Hbf[idx]  = f2bf(hv);
            }
        }
    }
}

// ---------------------------------------------------------------------------
// K4/K5: generic WMMA GEMM  out[8192, N] = A[8192,1024] @ Bt^T + bias
//   Bt is N-major (N x 1024 bf16). EPI: 0 = bf16 out, 1 = f32 out.
// Block tile 128 x 128; wave tile 32 x 64 (2x4); triple-buffered async LDS.
// N boundary handled by clamping staged rows (uniform async counts) and
// guarding epilogue stores.
// ---------------------------------------------------------------------------
template <int EPI>
__global__ void __launch_bounds__(256) wmma_gemm_kernel(
    const __bf16* __restrict__ A, const __bf16* __restrict__ Bt,
    const float* __restrict__ bias, void* __restrict__ outv, int Ntot)
{
    __shared__ alignas(16) __bf16 As[3 * 128 * LDA];
    __shared__ alignas(16) __bf16 Bs[3 * 128 * LDA];

    const int t    = threadIdx.x;
    const int lane = t & 31;
    const int wid  = t >> 5;
    const int wr   = wid & 3;     // 32-row group
    const int wc   = wid >> 2;    // 64-col group
    const int lr   = lane & 15;
    const int lh   = lane >> 4;

    const int m0 = blockIdx.x * 128;
    const int n0 = blockIdx.y * 128;

    // --- per-thread staging descriptors ---
    const int ar  = t >> 1;
    const int akc = (t & 1) << 4;
    const __bf16* agp = A + (size_t)(m0 + ar) * KK + akc;
    const unsigned al = lds_addr(&As[ar * LDA + akc]);

    const int br  = t >> 2;               // rows br and br+64 of the 128-row B tile
    const int bkc = (t & 3) << 3;
    int nA = n0 + br;       if (nA > Ntot - 1) nA = Ntot - 1;   // clamp: keeps
    int nB = n0 + br + 64;  if (nB > Ntot - 1) nB = Ntot - 1;   // async counts uniform
    const __bf16* bgA = Bt + (size_t)nA * KK + bkc;
    const __bf16* bgB = Bt + (size_t)nB * KK + bkc;
    const unsigned blA = lds_addr(&Bs[(br     ) * LDA + bkc]);
    const unsigned blB = lds_addr(&Bs[(br + 64) * LDA + bkc]);

    v8f acc[2][4] = {};

    auto stage = [&](int s) {
        const size_t  go = (size_t)s * 32;
        const unsigned bo = (unsigned)((s % 3) * (128 * LDA * 2));
        async_b128(al + bo,      agp + go);
        async_b128(al + bo + 16, agp + go + 8);
        async_b128(blA + bo, bgA + go);
        async_b128(blB + bo, bgB + go);
    };

    stage(0);
    const int NS = KK / 32;
    for (int s = 0; s < NS; ++s) {
        if (s + 1 < NS) { stage(s + 1); S_WAIT_ASYNC(4); }
        else            { S_WAIT_ASYNC(0); }
        __syncthreads();

        const __bf16* Ab = As + (s % 3) * (128 * LDA);
        const __bf16* Bb = Bs + (s % 3) * (128 * LDA);

        Frag a[2];
#pragma unroll
        for (int m = 0; m < 2; ++m) {
            const __bf16* p = Ab + (wr * 32 + m * 16 + lr) * LDA + lh * 8;
            a[m].h[0] = *(const v8bf*)p;
            a[m].h[1] = *(const v8bf*)(p + 16);
        }
#pragma unroll
        for (int n = 0; n < 4; ++n) {
            Frag b;
            const __bf16* q = Bb + (wc * 64 + n * 16 + lr) * LDA + lh * 8;
            b.h[0] = *(const v8bf*)q;
            b.h[1] = *(const v8bf*)(q + 16);
#pragma unroll
            for (int m = 0; m < 2; ++m)
                acc[m][n] = wmma_bf16(a[m], b, acc[m][n]);
        }
    }

#pragma unroll
    for (int n = 0; n < 4; ++n) {
        int col = n0 + wc * 64 + n * 16 + lr;
        if (col < Ntot) {
            float bv = bias[col];
#pragma unroll
            for (int m = 0; m < 2; ++m) {
                int rbase = m0 + wr * 32 + m * 16 + 8 * lh;
#pragma unroll
                for (int v = 0; v < 8; ++v) {
                    float val = acc[m][n][v] + bv;
                    size_t idx = (size_t)(rbase + v) * Ntot + col;
                    if (EPI == 0) ((__bf16*)outv)[idx] = f2bf(val);
                    else          ((float*)outv)[idx]  = val;
                }
            }
        }
    }
}

// ---------------------------------------------------------------------------
// Host launcher
// ---------------------------------------------------------------------------
extern "C" void kernel_launch(void* const* d_in, const int* in_sizes, int n_in,
                              void* d_out, int out_size, void* d_ws, size_t ws_size,
                              hipStream_t stream)
{
    const int*   S        = (const int*)d_in[0];
    const float* features = (const float*)d_in[1];
    const float* emb      = (const float*)d_in[2];
    const float* W_k      = (const float*)d_in[3];
    // d_in[4] = W_r : multiplied by h0 == 0, unused
    const float* b_l      = (const float*)d_in[5];
    const float* d1_w     = (const float*)d_in[6];
    const float* d1_b     = (const float*)d_in[7];
    const float* d2_w     = (const float*)d_in[8];
    const float* d2_b     = (const float*)d_in[9];

    float* logits = (float*)d_out;                       // 8192 x 12000
    float* outH   = logits + (size_t)BB * VV;            // 8192 x 1024
    float* outC   = outH + (size_t)BB * UU;              // 8192 x 1024

    char* ws = (char*)d_ws;
    __bf16* Xbf  = (__bf16*)(ws);                               // 16.78 MB
    __bf16* Hbf  = (__bf16*)(ws + 16777216);                    // 16.78 MB
    __bf16* X1bf = (__bf16*)(ws + 33554432);                    // 16.78 MB
    __bf16* WkT  = (__bf16*)(ws + 50331648);                    //  8.39 MB
    __bf16* d1T  = (__bf16*)(ws + 58720256);                    //  2.10 MB
    __bf16* d2T  = (__bf16*)(ws + 60817408);                    // 24.58 MB

    prep_x_kernel<<<(BB * KK) / 256, 256, 0, stream>>>(features, emb, S, Xbf);

    {
        dim3 blk(32, 8);
        dim3 gWk(4 * UU / 32, KK / 32);
        transpose_f32_bf16_kernel<<<gWk, blk, 0, stream>>>(W_k, WkT, KK, 4 * UU);
        dim3 gD1(UU / 32, UU / 32);
        transpose_f32_bf16_kernel<<<gD1, blk, 0, stream>>>(d1_w, d1T, UU, UU);
        dim3 gD2(VV / 32, UU / 32);
        transpose_f32_bf16_kernel<<<gD2, blk, 0, stream>>>(d2_w, d2T, UU, VV);
    }

    {
        dim3 grid(BB / 128, UU / 64);
        lstm_gemm1_kernel<<<grid, 256, 0, stream>>>(Xbf, WkT, b_l, outH, outC, Hbf);
    }
    {
        dim3 grid(BB / 128, UU / 128);
        wmma_gemm_kernel<0><<<grid, 256, 0, stream>>>(Hbf, d1T, d1_b, (void*)X1bf, UU);
    }
    {
        dim3 grid(BB / 128, (VV + 127) / 128);
        wmma_gemm_kernel<1><<<grid, 256, 0, stream>>>(X1bf, d2T, d2_b, (void*)logits, VV);
    }
}
